// RGCN_10471130268472
// MI455X (gfx1250) — matrix operand order repaired and checked by reading, not compiled, not run on hardware
//
#include <hip/hip_runtime.h>
#include <math.h>

#define N_NODES 50000
#define N_EDGES 200000
#define ETOT    (N_EDGES + N_NODES)     // 250000 incl. self loops
#define NREL    65
#define FDIM    128
#define HDIM    256
#define NGRAPH  1600
#define MDIM    64
#define MAXTILES 15686                  // ceil bound on sum_r ceil(E_r/16)
#define EPAD    (MAXTILES * 16)

typedef unsigned short bf16_t;          // storage type (host-safe)
typedef __attribute__((ext_vector_type(16))) __bf16 v16bf;
typedef __attribute__((ext_vector_type(8)))  __bf16 v8bf;
typedef __attribute__((ext_vector_type(8)))  float  v8f;

// ---------------- device helpers ----------------

__device__ __forceinline__ bf16_t f2bf(float f) {
    unsigned u = __float_as_uint(f);
    // round-to-nearest-even to bf16
    unsigned r = u + 0x7FFFu + ((u >> 16) & 1u);
    return (bf16_t)(r >> 16);
}

__device__ __forceinline__ v8f wmma_bf16(v16bf a, v16bf b, v8f c) {
    return __builtin_amdgcn_wmma_f32_16x16x32_bf16(
        /*neg_a=*/false, a, /*neg_b=*/false, b,
        /*c_mod=*/(short)0, c, /*reuse_a=*/false, /*reuse_b=*/false);
}

// A fragment: 16x32 bf16. lane -> row = lane&15; kb = (lane>>4)*8.
// elements 0..7  = A[row][k+kb .. k+kb+7]
// elements 8..15 = A[row][k+16+kb .. k+16+kb+7]
__device__ __forceinline__ v16bf load_a_frag(const bf16_t* rowPtr, int k, int lane) {
    int kb = (lane >> 4) * 8;
    union { v16bf v; v8bf h[2]; } u;
    u.h[0] = *reinterpret_cast<const v8bf*>(rowPtr + k + kb);
    u.h[1] = *reinterpret_cast<const v8bf*>(rowPtr + k + kb + 16);
    return u.v;
}

// B fragment: 32x16 bf16 with B stored TRANSPOSED as Bt[col][K].
// lane -> col = lane&15; kb = (lane>>4)*8; elements 0..7 = Bt[col][k+kb .. +7], 8..15 = Bt[col][k+16+kb .. +7]
__device__ __forceinline__ v16bf load_b_frag(const bf16_t* btColRow, int k, int lane) {
    int kb = (lane >> 4) * 8;
    union { v16bf v; v8bf h[2]; } u;
    u.h[0] = *reinterpret_cast<const v8bf*>(btColRow + k + kb);
    u.h[1] = *reinterpret_cast<const v8bf*>(btColRow + k + kb + 16);
    return u.v;
}

// ---------------- setup kernels ----------------

__global__ void fill_zero_u32(unsigned* p, long long n) {
    long long i = (long long)blockIdx.x * blockDim.x + threadIdx.x;
    if (i < n) p[i] = 0u;
}

__global__ void count_edges(const int* src, const int* dst, const int* et,
                            int* cnt, int* hist) {
    int e = blockIdx.x * blockDim.x + threadIdx.x;
    if (e >= ETOT) return;
    int d, r;
    if (e < N_EDGES) { d = dst[e]; r = et[e]; }
    else             { d = e - N_EDGES; r = 0; }
    atomicAdd(&cnt[d * NREL + r], 1);
    atomicAdd(&hist[r], 1);
}

__global__ void scan_hist(const int* hist, int* offP, int* cursor) {
    if (blockIdx.x == 0 && threadIdx.x == 0) {
        int off = 0;
        for (int r = 0; r < NREL; ++r) {
            offP[r] = off;
            cursor[r] = off;
            off += ((hist[r] + 15) >> 4) << 4;   // pad each segment to 16
        }
        offP[NREL] = off;
    }
}

__global__ void scatter_edges(const int* src, const int* dst, const int* et,
                              const int* cnt, int* cursor,
                              int* srcS, int* dstS, float* scaleS) {
    int e = blockIdx.x * blockDim.x + threadIdx.x;
    if (e >= ETOT) return;
    int s, d, r;
    if (e < N_EDGES) { s = src[e]; d = dst[e]; r = et[e]; }
    else             { s = e - N_EDGES; d = s; r = 0; }
    int pos = atomicAdd(&cursor[r], 1);
    srcS[pos] = s;
    dstS[pos] = d;
    scaleS[pos] = 1.0f / (float)cnt[d * NREL + r];
}

__global__ void cvt_bf16(const float* in, bf16_t* out, long long n) {
    long long i = (long long)blockIdx.x * blockDim.x + threadIdx.x;
    if (i < n) out[i] = f2bf(in[i]);
}

// W[b][k][h] (f32) -> Wt[b][h][k] (bf16)
__global__ void transpose_cvt(const float* in, bf16_t* out, int B, int K, int Hc) {
    long long n = (long long)B * K * Hc;
    long long i = (long long)blockIdx.x * blockDim.x + threadIdx.x;
    if (i >= n) return;
    int b = (int)(i / ((long long)K * Hc));
    int rem = (int)(i % ((long long)K * Hc));
    int k = rem / Hc;
    int h = rem % Hc;
    out[((long long)b * Hc + h) * K + k] = f2bf(in[i]);
}

__global__ void relu_cvt(const float* in, bf16_t* out, long long n) {
    long long i = (long long)blockIdx.x * blockDim.x + threadIdx.x;
    if (i < n) out[i] = f2bf(fmaxf(in[i], 0.0f));
}

// ---------------- WMMA GEMM kernels ----------------

// Out[row0..row0+16, :] = A(bf16,[rows,K]) @ Bt(bf16,[Hc,K])^T + bias
// block = 128 threads (4 waves); grid = (rows/16, Hc/64); each wave one 16x16 tile.
__global__ void gemm_bias(const bf16_t* __restrict__ A, const bf16_t* __restrict__ Bt,
                          const float* __restrict__ bias, float* __restrict__ Out,
                          int K, int Hc) {
    int lane = threadIdx.x & 31;
    int wave = threadIdx.x >> 5;
    int row0 = blockIdx.x * 16;
    int col0 = blockIdx.y * 64 + wave * 16;

    const bf16_t* aRow = A + (long long)(row0 + (lane & 15)) * K;
    const bf16_t* bRow = Bt + (long long)(col0 + (lane & 15)) * K;

    v8f acc = {0.f, 0.f, 0.f, 0.f, 0.f, 0.f, 0.f, 0.f};
    for (int k = 0; k < K; k += 32) {
        v16bf a = load_a_frag(aRow, k, lane);
        v16bf b = load_b_frag(bRow, k, lane);
        acc = wmma_bf16(a, b, acc);
    }
    int colw = col0 + (lane & 15);
    float bv = bias[colw];
    int mhalf = (lane >> 4) * 8;
#pragma unroll
    for (int j = 0; j < 8; ++j) {
        int row = row0 + j + mhalf;
        Out[(long long)row * Hc + colw] = acc[j] + bv;
    }
}

// One block = one 16-edge tile (single relation segment, padded).
// Hout[dst[e], :] += scale[e] * (X[src[e], :] @ W_r)
__global__ void edge_gemm(const int* __restrict__ srcS, const int* __restrict__ dstS,
                          const float* __restrict__ scaleS, const int* __restrict__ offP,
                          const bf16_t* __restrict__ X, const bf16_t* __restrict__ Wt,
                          float* __restrict__ Hout, int K, int Hc) {
    __shared__ int   s_src[16];
    __shared__ int   s_dst[16];
    __shared__ float s_scale[16];

    int tile0 = blockIdx.x * 16;
    if (tile0 >= offP[NREL]) return;

    int r = 0;
    for (int i = 1; i < NREL; ++i)
        if (tile0 >= offP[i]) r = i;

    if (threadIdx.x < 16) {
        s_src[threadIdx.x]   = srcS[tile0 + threadIdx.x];
        s_dst[threadIdx.x]   = dstS[tile0 + threadIdx.x];
        s_scale[threadIdx.x] = scaleS[tile0 + threadIdx.x];
    }
    __syncthreads();

    int lane = threadIdx.x & 31;
    int wave = threadIdx.x >> 5;

    const bf16_t* aRow = X + (long long)s_src[lane & 15] * K;
    const bf16_t* WtR  = Wt + (long long)r * Hc * K;

    const bf16_t* bRows[4];
#pragma unroll
    for (int t = 0; t < 4; ++t)
        bRows[t] = WtR + (long long)(wave * 64 + t * 16 + (lane & 15)) * K;

    v8f zero = {0.f, 0.f, 0.f, 0.f, 0.f, 0.f, 0.f, 0.f};
    v8f acc[4] = {zero, zero, zero, zero};

    for (int k = 0; k < K; k += 32) {
        v16bf a = load_a_frag(aRow, k, lane);
#pragma unroll
        for (int t = 0; t < 4; ++t) {
            v16bf b = load_b_frag(bRows[t], k, lane);
            acc[t] = wmma_bf16(a, b, acc[t]);
        }
    }

    int mhalf = (lane >> 4) * 8;
#pragma unroll
    for (int t = 0; t < 4; ++t) {
        int col = wave * 64 + t * 16 + (lane & 15);
#pragma unroll
        for (int j = 0; j < 8; ++j) {
            int m = j + mhalf;
            float v = acc[t][j] * s_scale[m];
            atomicAdd(&Hout[(long long)s_dst[m] * Hc + col], v);
        }
    }
}

// ---------------- pooling + MLP ----------------

// one wave per node: w = sigmoid(relu(h) . ws_w + ws_b); emb[batch] += w * relu(h)
__global__ void pool_kernel(const float* __restrict__ Hf, const int* __restrict__ batch,
                            const float* __restrict__ ws_w, const float* __restrict__ ws_b,
                            float* __restrict__ emb) {
    int wave = threadIdx.x >> 5;
    int lane = threadIdx.x & 31;
    int i = blockIdx.x * (blockDim.x >> 5) + wave;
    if (i >= N_NODES) return;

    float vals[8];
    float part = 0.f;
#pragma unroll
    for (int c8 = 0; c8 < 8; ++c8) {
        int c = lane + c8 * 32;
        float v = fmaxf(Hf[(long long)i * HDIM + c], 0.0f);
        vals[c8] = v;
        part += v * ws_w[c];
    }
#pragma unroll
    for (int off = 16; off >= 1; off >>= 1)
        part += __shfl_xor(part, off, 32);
    float w = 1.0f / (1.0f + expf(-(part + ws_b[0])));
    int b = batch[i];
#pragma unroll
    for (int c8 = 0; c8 < 8; ++c8)
        atomicAdd(&emb[(long long)b * HDIM + lane + c8 * 32], w * vals[c8]);
}

__global__ void mlp_fc(const float* __restrict__ in, const float* __restrict__ Wm,
                       const float* __restrict__ bm, float* __restrict__ out,
                       int rows, int Kdim, int Ncols, int doRelu) {
    int idx = blockIdx.x * blockDim.x + threadIdx.x;
    if (idx >= rows * Ncols) return;
    int g = idx / Ncols, j = idx % Ncols;
    float acc = bm[j];
    for (int k = 0; k < Kdim; ++k)
        acc += in[(long long)g * Kdim + k] * Wm[(long long)k * Ncols + j];
    out[idx] = doRelu ? fmaxf(acc, 0.0f) : acc;
}

__global__ void final_out(const float* __restrict__ z, const float* __restrict__ ow,
                          const float* __restrict__ ob, float* __restrict__ out) {
    int g = blockIdx.x * blockDim.x + threadIdx.x;
    if (g >= NGRAPH) return;
    float acc = ob[0];
    for (int k = 0; k < MDIM; ++k)
        acc += z[(long long)g * MDIM + k] * ow[k];
    out[g] = acc;
}

// ---------------- host launcher ----------------

static inline int blks(long long n, int b) { return (int)((n + b - 1) / b); }

extern "C" void kernel_launch(void* const* d_in, const int* in_sizes, int n_in,
                              void* d_out, int out_size, void* d_ws, size_t ws_size,
                              hipStream_t stream) {
    (void)in_sizes; (void)n_in; (void)out_size; (void)ws_size;

    const float* x     = (const float*)d_in[0];
    const int*   eidx  = (const int*)d_in[1];
    const int*   src   = eidx;
    const int*   dst   = eidx + N_EDGES;
    const int*   etype = (const int*)d_in[2];
    const int*   batch = (const int*)d_in[3];
    const float* W1    = (const float*)d_in[4];
    const float* root1 = (const float*)d_in[5];
    const float* b1    = (const float*)d_in[6];
    const float* W2    = (const float*)d_in[7];
    const float* root2 = (const float*)d_in[8];
    const float* b2    = (const float*)d_in[9];
    const float* ws_w  = (const float*)d_in[10];
    const float* ws_b  = (const float*)d_in[11];
    const float* m1w   = (const float*)d_in[12];
    const float* m1b   = (const float*)d_in[13];
    const float* m2w   = (const float*)d_in[14];
    const float* m2b   = (const float*)d_in[15];
    const float* m3w   = (const float*)d_in[16];
    const float* m3b   = (const float*)d_in[17];
    const float* ow    = (const float*)d_in[18];
    const float* ob    = (const float*)d_in[19];
    float* out = (float*)d_out;

    // workspace carving (256B aligned)
    size_t off = 0;
    char* base = (char*)d_ws;
    auto take = [&](size_t bytes) -> char* {
        char* p = base + off;
        off += (bytes + 255) & ~(size_t)255;
        return p;
    };
    int*    cnt    = (int*)take((size_t)N_NODES * NREL * 4);
    int*    hist   = (int*)take(NREL * 4);
    int*    offP   = (int*)take((NREL + 1) * 4);
    int*    cursor = (int*)take(NREL * 4);
    int*    srcS   = (int*)take((size_t)EPAD * 4);
    int*    dstS   = (int*)take((size_t)EPAD * 4);
    float*  scaleS = (float*)take((size_t)EPAD * 4);
    bf16_t* xb     = (bf16_t*)take((size_t)N_NODES * FDIM * 2);
    bf16_t* W1t    = (bf16_t*)take((size_t)NREL * FDIM * HDIM * 2);
    bf16_t* r1t    = (bf16_t*)take((size_t)FDIM * HDIM * 2);
    bf16_t* W2t    = (bf16_t*)take((size_t)NREL * HDIM * HDIM * 2);
    bf16_t* r2t    = (bf16_t*)take((size_t)HDIM * HDIM * 2);
    float*  hf     = (float*)take((size_t)N_NODES * HDIM * 4);
    bf16_t* h1b    = (bf16_t*)take((size_t)N_NODES * HDIM * 2);
    float*  emb    = (float*)take((size_t)NGRAPH * HDIM * 4);
    float*  z1     = (float*)take((size_t)NGRAPH * MDIM * 4);
    float*  z2     = (float*)take((size_t)NGRAPH * MDIM * 4);
    float*  z3     = (float*)take((size_t)NGRAPH * MDIM * 4);

    // 1. zero scratch that is accumulated into
    fill_zero_u32<<<blks((long long)N_NODES * NREL, 256), 256, 0, stream>>>((unsigned*)cnt, (long long)N_NODES * NREL);
    fill_zero_u32<<<1, 256, 0, stream>>>((unsigned*)hist, NREL);
    fill_zero_u32<<<blks(EPAD, 256), 256, 0, stream>>>((unsigned*)srcS, EPAD);
    fill_zero_u32<<<blks(EPAD, 256), 256, 0, stream>>>((unsigned*)dstS, EPAD);
    fill_zero_u32<<<blks(EPAD, 256), 256, 0, stream>>>((unsigned*)scaleS, EPAD);
    fill_zero_u32<<<blks((long long)NGRAPH * HDIM, 256), 256, 0, stream>>>((unsigned*)emb, (long long)NGRAPH * HDIM);

    // 2. counting sort of edges by relation (with per-(dst,rel) counts for the mean)
    count_edges<<<blks(ETOT, 256), 256, 0, stream>>>(src, dst, etype, cnt, hist);
    scan_hist<<<1, 1, 0, stream>>>(hist, offP, cursor);
    scatter_edges<<<blks(ETOT, 256), 256, 0, stream>>>(src, dst, etype, cnt, cursor, srcS, dstS, scaleS);

    // 3. bf16 conversions (weights transposed to [Hc,K] for contiguous B fragments)
    cvt_bf16<<<blks((long long)N_NODES * FDIM, 256), 256, 0, stream>>>(x, xb, (long long)N_NODES * FDIM);
    transpose_cvt<<<blks((long long)NREL * FDIM * HDIM, 256), 256, 0, stream>>>(W1, W1t, NREL, FDIM, HDIM);
    transpose_cvt<<<blks((long long)FDIM * HDIM, 256), 256, 0, stream>>>(root1, r1t, 1, FDIM, HDIM);
    transpose_cvt<<<blks((long long)NREL * HDIM * HDIM, 256), 256, 0, stream>>>(W2, W2t, NREL, HDIM, HDIM);
    transpose_cvt<<<blks((long long)HDIM * HDIM, 256), 256, 0, stream>>>(root2, r2t, 1, HDIM, HDIM);

    // 4. layer 1: root GEMM (stores) then per-relation edge GEMM (scaled atomics)
    gemm_bias<<<dim3(N_NODES / 16, HDIM / 64), 128, 0, stream>>>(xb, r1t, b1, hf, FDIM, HDIM);
    edge_gemm<<<MAXTILES, 128, 0, stream>>>(srcS, dstS, scaleS, offP, xb, W1t, hf, FDIM, HDIM);
    relu_cvt<<<blks((long long)N_NODES * HDIM, 256), 256, 0, stream>>>(hf, h1b, (long long)N_NODES * HDIM);

    // 5. layer 2 (hf reused as f32 accumulator)
    gemm_bias<<<dim3(N_NODES / 16, HDIM / 64), 128, 0, stream>>>(h1b, r2t, b2, hf, HDIM, HDIM);
    edge_gemm<<<MAXTILES, 128, 0, stream>>>(srcS, dstS, scaleS, offP, h1b, W2t, hf, HDIM, HDIM);

    // 6. weighted-sum pooling (relu fused into the read)
    pool_kernel<<<blks((long long)N_NODES * 32, 256), 256, 0, stream>>>(hf, batch, ws_w, ws_b, emb);

    // 7. MLP head
    mlp_fc<<<blks((long long)NGRAPH * MDIM, 256), 256, 0, stream>>>(emb, m1w, m1b, z1, NGRAPH, HDIM, MDIM, 1);
    mlp_fc<<<blks((long long)NGRAPH * MDIM, 256), 256, 0, stream>>>(z1, m2w, m2b, z2, NGRAPH, MDIM, MDIM, 1);
    mlp_fc<<<blks((long long)NGRAPH * MDIM, 256), 256, 0, stream>>>(z2, m3w, m3b, z3, NGRAPH, MDIM, MDIM, 0);
    final_out<<<blks(NGRAPH, 256), 256, 0, stream>>>(z3, ow, ob, out);
}
